// LMM_13134009991698
// MI455X (gfx1250) — matrix operand order repaired and checked by reading, not compiled, not run on hardware
//
#include <hip/hip_runtime.h>

typedef __bf16 bf16_t;
typedef __attribute__((ext_vector_type(16))) __bf16 v16bf;
typedef __attribute__((ext_vector_type(8)))  float  v8f;
typedef unsigned int v4u __attribute__((ext_vector_type(4)));
typedef int v8i __attribute__((ext_vector_type(8)));
typedef int v4i __attribute__((ext_vector_type(4)));

constexpr int kB = 8;
constexpr int kL = 2048;
constexpr int kM = 4096;
constexpr int kD = 512;
constexpr int kTopK = 5;
constexpr int KT  = 32;            // K per v_wmma_f32_16x16x32_bf16
constexpr int NKF = kD / KT;       // 16 K-fragments to cover D=512
constexpr int WAVES = 4;           // waves per block in kernel 2
constexpr int NTILES = kM / 16;    // 256 B-tiles (16 memory rows each)

// ---------------------------------------------------------------------------
// Branchless sorted top-5 insertion (t[0] >= t[1] >= ... >= t[4])
// ---------------------------------------------------------------------------
__device__ __forceinline__ void topk_insert(float v, int idx, float t[5], int ti[5]) {
  if (v <= t[4]) return;
  const bool g0 = v > t[0], g1 = v > t[1], g2 = v > t[2], g3 = v > t[3];
  t[4]  = g3 ? t[3]  : v;              ti[4] = g3 ? ti[3] : idx;
  t[3]  = g3 ? (g2 ? t[2] : v) : t[3]; ti[3] = g3 ? (g2 ? ti[2] : idx) : ti[3];
  t[2]  = g2 ? (g1 ? t[1] : v) : t[2]; ti[2] = g2 ? (g1 ? ti[1] : idx) : ti[2];
  t[1]  = g1 ? (g0 ? t[0] : v) : t[1]; ti[1] = g1 ? (g0 ? ti[0] : idx) : ti[1];
  t[0]  = g0 ? v : t[0];               ti[0] = g0 ? idx : ti[0];
}

__device__ __forceinline__ void lds_frag(v16bf& dst, const bf16_t* p) {
  union { v16bf v; uint4 q[2]; } u;
  u.q[0] = *(const uint4*)(p);         // ds_load_b128
  u.q[1] = *(const uint4*)(p + 16);    // ds_load_b128
  dst = u.v;
}

// ---------------------------------------------------------------------------
// Kernel 1: L2-normalize rows of encoded & memory, emit bf16 into workspace.
// One wave per row of 512 floats; 8 rows per 256-thread block.
// ---------------------------------------------------------------------------
__global__ __launch_bounds__(256) void norm_cvt_kernel(
    const float* __restrict__ enc, const float* __restrict__ mem,
    bf16_t* __restrict__ encn, bf16_t* __restrict__ memn) {
  const int lane = threadIdx.x & 31;
  int row = blockIdx.x * (blockDim.x >> 5) + (threadIdx.x >> 5);
  const int rowsEnc = kB * kL;
  const float* src;
  bf16_t* dst;
  if (row < rowsEnc) {
    src = enc + (size_t)row * kD;
    dst = encn + (size_t)row * kD;
  } else {
    row -= rowsEnc;
    src = mem + (size_t)row * kD;
    dst = memn + (size_t)row * kD;
  }
  const int base = lane * 16;       // 16 contiguous elements per lane
  union { float4 v4[4]; float f[16]; } x;
  float ss = 0.f;
#pragma unroll
  for (int q = 0; q < 4; ++q) {
    x.v4[q] = *(const float4*)(src + base + q * 4);
    ss += x.v4[q].x * x.v4[q].x + x.v4[q].y * x.v4[q].y +
          x.v4[q].z * x.v4[q].z + x.v4[q].w * x.v4[q].w;
  }
#pragma unroll
  for (int off = 16; off > 0; off >>= 1) ss += __shfl_xor(ss, off, 32);
  const float scale = 1.0f / fmaxf(sqrtf(ss), 1e-12f);
  union { bf16_t h[16]; uint4 q2[2]; } o;
#pragma unroll
  for (int i = 0; i < 16; ++i) o.h[i] = (bf16_t)(x.f[i] * scale);
  *(uint4*)(dst + base)     = o.q2[0];
  *(uint4*)(dst + base + 8) = o.q2[1];
}

// ---------------------------------------------------------------------------
// Kernel 2: TDM-fed bf16-WMMA cosine-sim GEMM + streaming top-5 + gather.
// Block = 128 threads (4 waves) owns (batch b, 64 encoded rows); each wave
// owns a 16-row tile and keeps its 16 A-fragments (16x512 bf16) in VGPRs.
// All 4 waves sweep the full M=4096 together: wave 0 drives a double-buffered
// tensor_load_to_lds stream of 16-row B-tiles (16 KB each, TENSORcnt-tracked);
// the compute loop is a sched_barrier-pinned ds_load/WMMA pipeline.
// ---------------------------------------------------------------------------
__global__ __launch_bounds__(128) void simtopk_kernel(
    const float* __restrict__ enc, const float* __restrict__ mem,
    const bf16_t* __restrict__ encn, const bf16_t* __restrict__ memn,
    float* __restrict__ out) {
  __shared__ bf16_t bBuf[2][16 * kD];          // 2 x 16KB B-tile double buffer
  __shared__ float  simLds[WAVES][16][17];     // per-wave 16x16 C staging (+pad)
  __shared__ int    fIdx[WAVES][16][kTopK];    // per-wave final top-5 indices

  const int b    = blockIdx.x >> 5;            // 32 row-blocks (64 rows) / batch
  const int r0   = (blockIdx.x & 31) * 64 + (threadIdx.x >> 5) * 16;
  const int wave = threadIdx.x >> 5;
  const int lane = threadIdx.x & 31;
  const int half = lane >> 4;                  // 0: lanes 0-15, 1: lanes 16-31
  const int l15  = lane & 15;

  // ---- A fragments: rows r0..r0+15, full K=512, ISA 16-bit A striping ----
  // lanes 0-15 hold K {k0+0..7, k0+16..23}; lanes 16-31 hold {k0+8..15, k0+24..31}
  const bf16_t* encRow = encn + ((size_t)(b * kL + r0 + l15)) * kD;
  v16bf A[NKF];
  {
    const int aOff = half * 8;
#pragma unroll
    for (int kf = 0; kf < NKF; ++kf) {
      union { v16bf v; uint4 q[2]; } u;
      u.q[0] = *(const uint4*)(encRow + kf * KT + aOff);
      u.q[1] = *(const uint4*)(encRow + kf * KT + aOff + 16);
      A[kf] = u.v;
    }
  }

  // per-lane running top-5 (2 lanes per row: lane r covers cols 0-7 of each
  // tile, lane r+16 covers cols 8-15)
  float tv[kTopK] = {-3.4e38f, -3.4e38f, -3.4e38f, -3.4e38f, -3.4e38f};
  int   ti[kTopK] = {0, 0, 0, 0, 0};

  const bf16_t* memBase = memn + (size_t)b * kM * kD;

  // ---- TDM: DMA one 16-row x 512-col bf16 tile (16 KB) into LDS ----------
  // D# treats each tile as its own 2D tensor: data_size=2B, dims 512x16,
  // row stride 512; groups 2/3 zero (2D); type=2 ("image"), count=1.
  auto issue_tile = [&](int t, int buf) {
    const uint64_t ga = (uint64_t)(uintptr_t)(memBase + (size_t)t * 16 * kD);
    const unsigned ldsAddr =
        (unsigned)(uintptr_t)&bBuf[buf][0];     // low 32 bits = LDS byte offset
    v4u g0;
    g0[0] = 1u;                                 // count=1, user mode
    g0[1] = ldsAddr;                            // lds_addr
    g0[2] = (unsigned)ga;                       // global_addr[31:0]
    g0[3] = ((unsigned)(ga >> 32) & 0x01FFFFFFu) | (2u << 30);  // [56:32]|type=2
    v8i g1;
    g1[0] = (1 << 16);                          // data_size = 2 bytes
    g1[1] = (512 << 16);                        // tensor_dim0 = 512 (bits 79:48)
    g1[2] = (16 << 16);                         // tensor_dim1 = 16  (bits 111:80)
    g1[3] = (512 << 16);                        // tile_dim0 = 512   (bits 127:112)
    g1[4] = 16;                                 // tile_dim1 = 16    (bits 143:128)
    g1[5] = 512;                                // tensor_dim0_stride = 512
    g1[6] = 0;
    g1[7] = 0;
    const v4i gz4 = {0, 0, 0, 0};               // 2D: groups 2/3 unused
    const v8i gz8 = {0, 0, 0, 0, 0, 0, 0, 0};   // extra descriptor words unused
    __builtin_amdgcn_tensor_load_to_lds(g0, g1, gz4, gz4, gz8, 0);
  };

  if (wave == 0) {                              // prime double buffer
    issue_tile(0, 0);
    issue_tile(1, 1);
  }

  for (int t = 0; t < NTILES; ++t) {
    if (wave == 0) {
      if (t + 1 < NTILES) __builtin_amdgcn_s_wait_tensorcnt(1);  // tile t done
      else                __builtin_amdgcn_s_wait_tensorcnt(0);
    }
    __syncthreads();   // buffer (t&1) ready for all waves

    const int n0 = t * 16;
    // B striping: lane holds memrow (col) l15, K {kOff..kOff+7, kOff+16..kOff+23} per frag (kOff = half*8)
    const bf16_t* bp = &bBuf[t & 1][(size_t)l15 * kD + half * 8];

    // Distance-2 ds pipeline, pinned with sched_barrier(0) so the machine
    // scheduler cannot sink the prefetch loads back onto their consumer:
    // before each wmma only s_wait_dscnt 0x4 is needed (2 newer pairs in
    // flight), hiding LDS latency behind two WMMA issues.
    v16bf Bb[4];
    lds_frag(Bb[0], bp);
    lds_frag(Bb[1], bp + KT);
    v8f C = {};
#pragma unroll
    for (int kf = 0; kf < NKF; ++kf) {
      if (kf + 2 < NKF) lds_frag(Bb[(kf + 2) & 3], bp + (kf + 2) * KT);
      __builtin_amdgcn_sched_barrier(0);
      C = __builtin_amdgcn_wmma_f32_16x16x32_bf16(
          /*neg_a=*/false, A[kf], /*neg_b=*/false, Bb[kf & 3],
          /*c_mod=*/(short)0, C, /*reuse_a=*/false, /*reuse_b=*/false);
      __builtin_amdgcn_sched_barrier(0);
    }

    // stage C tile through per-wave-private LDS (C layout: lane<16 -> rows 0-7,
    // lane>=16 -> rows 8-15; col = lane&15). DS ops from one wave complete in
    // order; wave_barrier stops the compiler from reordering them.
#pragma unroll
    for (int i = 0; i < 8; ++i) simLds[wave][half * 8 + i][l15] = C[i];
    __builtin_amdgcn_wave_barrier();
#pragma unroll
    for (int c = 0; c < 8; ++c) {
      const int col = half * 8 + c;
      topk_insert(simLds[wave][l15][col], n0 + col, tv, ti);
    }
    __builtin_amdgcn_wave_barrier();

    __syncthreads();   // all waves done reading buffer (t&1)
    if (wave == 0 && t + 2 < NTILES) issue_tile(t + 2, t & 1);
  }

  // merge the two column-halves of each row (lane r <-> lane r+16)
  {
    float ov[kTopK]; int oi[kTopK];
#pragma unroll
    for (int j = 0; j < kTopK; ++j) {
      ov[j] = __shfl_xor(tv[j], 16, 32);
      oi[j] = __shfl_xor(ti[j], 16, 32);
    }
#pragma unroll
    for (int j = 0; j < kTopK; ++j) topk_insert(ov[j], oi[j], tv, ti);
  }
  if (lane < 16) {
#pragma unroll
    for (int j = 0; j < kTopK; ++j) fIdx[wave][lane][j] = ti[j];
  }
  __builtin_amdgcn_wave_barrier();   // per-wave private region, DS in-order

  // output: out = enc + mean(top-5 unnormalized memory rows), 16 rows / wave,
  // each lane covers 16 consecutive floats of each row.
  const float* memF = mem + (size_t)b * kM * kD;
  const int d0 = lane * 16;
  for (int r = 0; r < 16; ++r) {
    int id[kTopK];
#pragma unroll
    for (int j = 0; j < kTopK; ++j) id[j] = fIdx[wave][r][j];
    const size_t rowOff = ((size_t)(b * kL + r0 + r)) * kD + d0;
    const float* encP = enc + rowOff;
    float*       outP = out + rowOff;
#pragma unroll
    for (int u2 = 0; u2 < 4; ++u2) {
      const float4 e = *(const float4*)(encP + u2 * 4);
      float4 s = make_float4(0.f, 0.f, 0.f, 0.f);
#pragma unroll
      for (int j = 0; j < kTopK; ++j) {
        const float4 m4 = *(const float4*)(memF + (size_t)id[j] * kD + d0 + u2 * 4);
        s.x += m4.x; s.y += m4.y; s.z += m4.z; s.w += m4.w;
      }
      float4 rr;
      rr.x = e.x + s.x * 0.2f;
      rr.y = e.y + s.y * 0.2f;
      rr.z = e.z + s.z * 0.2f;
      rr.w = e.w + s.w * 0.2f;
      *(float4*)(outP + u2 * 4) = rr;
    }
  }
}

// ---------------------------------------------------------------------------
extern "C" void kernel_launch(void* const* d_in, const int* in_sizes, int n_in,
                              void* d_out, int out_size, void* d_ws, size_t ws_size,
                              hipStream_t stream) {
  const float* enc = (const float*)d_in[0];   // [8,2048,512] f32
  const float* mem = (const float*)d_in[1];   // [8,4096,512] f32
  float* out = (float*)d_out;                 // [8,2048,512] f32

  // workspace: normalized bf16 copies (16.8 MB + 33.6 MB)
  bf16_t* encn = (bf16_t*)d_ws;
  bf16_t* memn = encn + (size_t)kB * kL * kD;

  // Kernel 1: 49152 rows total, 8 rows (waves) per 256-thread block
  const int rowsTotal = kB * (kL + kM);
  norm_cvt_kernel<<<rowsTotal / 8, 256, 0, stream>>>(enc, mem, encn, memn);

  // Kernel 2: one block per (batch, 64-row tile) = 8 * 32 = 256 blocks
  simtopk_kernel<<<kB * (kL / 64), 128, 0, stream>>>(enc, mem, encn, memn, out);
}
